// MultiheadAttention_34720515620893
// MI455X (gfx1250) — matrix operand order repaired and checked by reading, not compiled, not run on hardware
//
#include <hip/hip_runtime.h>
#include <hip/hip_fp16.h>
#include <stdint.h>

typedef _Float16 half_t;
typedef __attribute__((ext_vector_type(16))) _Float16 v16h;
typedef __attribute__((ext_vector_type(8)))  _Float16 v8h;
typedef __attribute__((ext_vector_type(8)))  float    v8f;

#define S_LEN 2048
#define BATCH 4
#define EMB   1024
#define NH    16
#define HD    64

static __device__ __forceinline__ int lane_id() { return threadIdx.x & 31; }

// Generic shared-memory pointer -> 32-bit LDS byte offset (ISA: LDS aperture
// address low 32 bits are the LDS address).
static __device__ __forceinline__ uint32_t lds_addr32(const void* p) {
  return (uint32_t)(uintptr_t)p;
}

// ---- WMMA fragment loaders (ISA 7.12.2 layouts, wave32) ----

// A fragment 16x32 f16, A row-major with leading dim lda (halves).
// lane<16: row=lane,   k = {k0..k0+7, k0+16..k0+23}
// lane>=16: row=lane-16, k = {k0+8..k0+15, k0+24..k0+31}
static __device__ __forceinline__ v16h load_frag_a_f16(const half_t* A, int lda) {
  const int l = lane_id();
  const int row = l & 15;
  const half_t* p = A + (size_t)row * lda + ((l & 16) ? 8 : 0);
  v8h lo = *(const v8h*)(p);
  v8h hi = *(const v8h*)(p + 16);
  v16h f;
#pragma unroll
  for (int i = 0; i < 8; ++i) { f[i] = lo[i]; f[8 + i] = hi[i]; }
  return f;
}

// Same A layout but source is f32 in LDS (convert on the fly).
static __device__ __forceinline__ v16h load_frag_a_lds_f32(const float* A, int lda) {
  const int l = lane_id();
  const int row = l & 15;
  const float* p = A + row * lda + ((l & 16) ? 8 : 0);
  v16h f;
#pragma unroll
  for (int i = 0; i < 8; ++i) { f[i] = (half_t)p[i]; f[8 + i] = (half_t)p[16 + i]; }
  return f;
}

// B fragment 32x16 f16 from Bt stored [N][K] row-major (ldb in halves).
// lane<16: col=lane,   k = {k0..k0+7, k0+16..k0+23}
// lane>=16: col=lane-16, k = {k0+8..k0+15, k0+24..k0+31}
// -> two 16-byte loads per lane (works for global or LDS pointers).
static __device__ __forceinline__ v16h load_frag_b_f16(const half_t* Bt, int ldb) {
  const int l = lane_id();
  const int col = l & 15;
  const half_t* p = Bt + (size_t)col * ldb + ((l & 16) ? 8 : 0);
  v8h lo = *(const v8h*)(p);
  v8h hi = *(const v8h*)(p + 16);
  v16h f;
#pragma unroll
  for (int i = 0; i < 8; ++i) { f[i] = lo[i]; f[8 + i] = hi[i]; }
  return f;
}

static __device__ __forceinline__ v8f wmma_f16(v16h a, v16h b, v8f c) {
  return __builtin_amdgcn_wmma_f32_16x16x32_f16(false, a, false, b, (short)0, c,
                                                false, false);
}

// ---- f32 -> f16 cast ----
__global__ void f32_to_f16_kernel(const float* __restrict__ src,
                                  half_t* __restrict__ dst, int n) {
  int i = blockIdx.x * blockDim.x + threadIdx.x;
  const int stride = gridDim.x * blockDim.x;
  for (; i < n; i += stride) dst[i] = (half_t)src[i];
}

// ---- QKV projection: C[8192,3072] = xh[8192,1024] * Wt, scatter Q/K/Vt ----
__global__ __launch_bounds__(256) void qkv_gemm_kernel(
    const half_t* __restrict__ xh, const half_t* __restrict__ wh,
    const float* __restrict__ bias, half_t* __restrict__ qws,
    half_t* __restrict__ kws, half_t* __restrict__ vtws) {
  const int w = threadIdx.x >> 5;
  const int l = threadIdx.x & 31;
  const int mbase = blockIdx.y * 32 + (w & 1) * 16;
  const int nbase = blockIdx.x * 256 + (w >> 1) * 64;

  v8f acc[4] = {};
  for (int k0 = 0; k0 < 1024; k0 += 32) {
    v16h a = load_frag_a_f16(xh + (size_t)mbase * 1024 + k0, 1024);
    if (k0 + 32 < 1024)
      __builtin_prefetch(xh + (size_t)mbase * 1024 + k0 + 32, 0, 1);
#pragma unroll
    for (int t = 0; t < 4; ++t) {
      v16h b = load_frag_b_f16(wh + (size_t)(nbase + t * 16) * 1024 + k0, 1024);
      acc[t] = wmma_f16(a, b, acc[t]);
    }
  }

  const int col = l & 15;
  const int r0 = (l & 16) ? 8 : 0;
#pragma unroll
  for (int t = 0; t < 4; ++t) {
#pragma unroll
    for (int i = 0; i < 8; ++i) {
      const int m = mbase + r0 + i;       // m = s*B + b
      const int n = nbase + t * 16 + col; // column in [0,3E)
      const float v = acc[t][i] + bias[n];
      const int s = m >> 2, b = m & 3;
      if (n < EMB) {                       // Q, pre-scaled by head_dim^-0.5
        const int h = n >> 6, d = n & 63;
        qws[(((size_t)(b * NH + h) * S_LEN) + s) * HD + d] = (half_t)(v * 0.125f);
      } else if (n < 2 * EMB) {            // K: [bh][s][d]
        const int e = n - EMB, h = e >> 6, d = e & 63;
        kws[(((size_t)(b * NH + h) * S_LEN) + s) * HD + d] = (half_t)v;
      } else {                             // V transposed: [bh][d][s]
        const int e = n - 2 * EMB, h = e >> 6, d = e & 63;
        vtws[(((size_t)(b * NH + h) * HD) + d) * S_LEN + s] = (half_t)v;
      }
    }
  }
}

// ---- fused attention: scores -> softmax -> attn weights out + P@V ----
// grid (qtile=64, bh=64), 8 waves.
// LDS: 32x2048 f32 scores (256KB) + reduction scratch + 32KB K-panel stage
//      (async-copied) = ~289KB of the 320KB WGP LDS.
__global__ __launch_bounds__(256) void attn_kernel(
    const half_t* __restrict__ qws, const half_t* __restrict__ kws,
    const half_t* __restrict__ vtws, half_t* __restrict__ ctx,
    float* __restrict__ attn_w) {
  extern __shared__ float smem[];
  float* sc = smem;                           // [32][2048]
  float* red = smem + 32 * 2048;              // [256]
  float* rowred = red + 256;                  // [32]
  half_t* kstage = (half_t*)(rowred + 32);    // [256][64] f16 = 32KB

  const int qt = blockIdx.x;
  const int bh = blockIdx.y;
  const int qbase = qt * 32;
  const int w = threadIdx.x >> 5;
  const int l = threadIdx.x & 31;

  const half_t* qh = qws + (size_t)bh * S_LEN * HD;
  const half_t* kh = kws + (size_t)bh * S_LEN * HD;
  const half_t* vt = vtws + (size_t)bh * HD * S_LEN;

  // Phase 1: scores[32][2048] = Q_tile @ K^T.
  // K streamed through LDS in 8 blocks of 256 keys via async-to-LDS DMA;
  // B-fragments then come from LDS, shared by all 8 waves.
  {
    v16h a00 = load_frag_a_f16(qh + (size_t)qbase * HD + 0, HD);
    v16h a01 = load_frag_a_f16(qh + (size_t)qbase * HD + 32, HD);
    v16h a10 = load_frag_a_f16(qh + (size_t)(qbase + 16) * HD + 0, HD);
    v16h a11 = load_frag_a_f16(qh + (size_t)(qbase + 16) * HD + 32, HD);
    const int col = l & 15;
    const int r0 = (l & 16) ? 8 : 0;
    const uint32_t ldsbase = lds_addr32(kstage);

    for (int nb = 0; nb < 8; ++nb) {
      const int kbase = nb * 256;
      // Cooperative async copy: 256 keys x 64 halves = 32KB, b128 per issue.
      {
        const uint64_t g =
            (uint64_t)(uintptr_t)(kh + (size_t)kbase * HD);
#pragma unroll
        for (int j = 0; j < 8; ++j) {
          const uint32_t off = (uint32_t)(j * 4096 + threadIdx.x * 16);
          asm volatile("global_load_async_to_lds_b128 %0, %1, off"
                       :: "v"(ldsbase + off), "v"(g + (uint64_t)off)
                       : "memory");
        }
        asm volatile("s_wait_asynccnt 0x0" ::: "memory");
      }
      __syncthreads();

      // Each wave: 2 N-tiles of this 256-key block (16 tiles / 8 waves).
#pragma unroll
      for (int nt = 0; nt < 2; ++nt) {
        const int ncl = (w * 2 + nt) * 16;  // local column within block
        v16h b0 = load_frag_b_f16(kstage + (size_t)ncl * HD + 0, HD);
        v16h b1 = load_frag_b_f16(kstage + (size_t)ncl * HD + 32, HD);
        v8f acc0 = {}, acc1 = {};
        acc0 = wmma_f16(a00, b0, acc0);
        acc0 = wmma_f16(a01, b1, acc0);
        acc1 = wmma_f16(a10, b0, acc1);
        acc1 = wmma_f16(a11, b1, acc1);
        const int nc = kbase + ncl;
#pragma unroll
        for (int i = 0; i < 8; ++i) {
          sc[(r0 + i) * 2048 + nc + col] = acc0[i];
          sc[(16 + r0 + i) * 2048 + nc + col] = acc1[i];
        }
      }
      __syncthreads();  // protect kstage before next block's DMA
    }
  }
  __syncthreads();

  // Phase 2: exact softmax per row (8 threads x 256 elems per row),
  // write normalized weights to global, keep normalized f32 in LDS for AV.
  {
    const int row = threadIdx.x >> 3;
    const int chunk = threadIdx.x & 7;
    float* r = sc + row * 2048 + chunk * 256;

    float m = -3.4e38f;
    for (int i = 0; i < 256; ++i) m = fmaxf(m, r[i]);
    red[threadIdx.x] = m;
    __syncthreads();
    if (chunk == 0) {
      float mm = red[row * 8];
#pragma unroll
      for (int j = 1; j < 8; ++j) mm = fmaxf(mm, red[row * 8 + j]);
      rowred[row] = mm;
    }
    __syncthreads();
    const float rm = rowred[row];

    float s = 0.f;
    for (int i = 0; i < 256; ++i) {
      const float e = __expf(r[i] - rm);
      r[i] = e;
      s += e;
    }
    red[threadIdx.x] = s;
    __syncthreads();
    if (chunk == 0) {
      float ss = 0.f;
#pragma unroll
      for (int j = 0; j < 8; ++j) ss += red[row * 8 + j];
      rowred[row] = ss;
    }
    __syncthreads();
    const float inv = 1.0f / rowred[row];

    float* g = attn_w + ((size_t)bh * S_LEN + (qbase + row)) * S_LEN + chunk * 256;
    for (int i = 0; i < 256; i += 4) {
      float4 p;
      p.x = r[i] * inv;     p.y = r[i + 1] * inv;
      p.z = r[i + 2] * inv; p.w = r[i + 3] * inv;
      r[i] = p.x; r[i + 1] = p.y; r[i + 2] = p.z; r[i + 3] = p.w;
      *(float4*)(g + i) = p;
    }
  }
  __syncthreads();

  // Phase 3: ctx[32][64] = P @ V ; one 16x16 tile per wave, K-loop of 64.
  {
    const int mt = w & 1;
    const int nt = w >> 1; // 0..3
    v8f acc = {};
    for (int ks = 0; ks < 64; ++ks) {
      v16h a = load_frag_a_lds_f32(sc + (mt * 16) * 2048 + ks * 32, 2048);
      v16h b = load_frag_b_f16(vt + (size_t)(nt * 16) * S_LEN + ks * 32, S_LEN);
      acc = wmma_f16(a, b, acc);
    }
    const int col = l & 15;
    const int r0 = (l & 16) ? 8 : 0;
    const int b_ = bh >> 4, h = bh & 15;
#pragma unroll
    for (int i = 0; i < 8; ++i) {
      const int sq = qbase + mt * 16 + r0 + i;
      const int e = h * 64 + nt * 16 + col;
      ctx[((size_t)sq * BATCH + b_) * EMB + e] = (half_t)acc[i];
    }
  }
}

// ---- output projection: out[8192,1024] = ctx @ Wout^T + bias (f32 out) ----
__global__ __launch_bounds__(256) void out_gemm_kernel(
    const half_t* __restrict__ ctx, const half_t* __restrict__ wout,
    const float* __restrict__ bias, float* __restrict__ out) {
  const int w = threadIdx.x >> 5;
  const int l = threadIdx.x & 31;
  const int mbase = blockIdx.y * 32 + (w & 1) * 16;
  const int nbase = blockIdx.x * 256 + (w >> 1) * 64;

  v8f acc[4] = {};
  for (int k0 = 0; k0 < 1024; k0 += 32) {
    v16h a = load_frag_a_f16(ctx + (size_t)mbase * 1024 + k0, 1024);
    if (k0 + 32 < 1024)
      __builtin_prefetch(ctx + (size_t)mbase * 1024 + k0 + 32, 0, 1);
#pragma unroll
    for (int t = 0; t < 4; ++t) {
      v16h b = load_frag_b_f16(wout + (size_t)(nbase + t * 16) * 1024 + k0, 1024);
      acc[t] = wmma_f16(a, b, acc[t]);
    }
  }
  const int col = l & 15;
  const int r0 = (l & 16) ? 8 : 0;
#pragma unroll
  for (int t = 0; t < 4; ++t) {
#pragma unroll
    for (int i = 0; i < 8; ++i) {
      const int m = mbase + r0 + i;
      const int n = nbase + t * 16 + col;
      out[(size_t)m * 1024 + n] = acc[t][i] + bias[n];
    }
  }
}

extern "C" void kernel_launch(void* const* d_in, const int* in_sizes, int n_in,
                              void* d_out, int out_size, void* d_ws, size_t ws_size,
                              hipStream_t stream) {
  (void)in_sizes; (void)n_in; (void)out_size; (void)ws_size;
  const float* x  = (const float*)d_in[0];
  const float* wi = (const float*)d_in[1];
  const float* bi = (const float*)d_in[2];
  const float* wo = (const float*)d_in[3];
  const float* bo = (const float*)d_in[4];
  // d_in[5] = num_heads (hardcoded NH=16)

  half_t* ws = (half_t*)d_ws;
  half_t* xh   = ws;                                   // 8192*1024
  half_t* wih  = xh   + (size_t)8192 * 1024;           // 3072*1024
  half_t* woh  = wih  + (size_t)3072 * 1024;           // 1024*1024
  half_t* qws  = woh  + (size_t)1024 * 1024;           // 64*2048*64
  half_t* kws  = qws  + (size_t)64 * 2048 * 64;        // 64*2048*64
  half_t* vtws = kws  + (size_t)64 * 2048 * 64;        // 64*2048*64 (transposed)
  half_t* ctx  = vtws + (size_t)64 * 2048 * 64;        // 8192*1024

  float* attn_out = (float*)d_out;                         // [2048,4,1024]
  float* attn_w   = attn_out + (size_t)2048 * 4 * 1024;    // [64,2048,2048]

  f32_to_f16_kernel<<<dim3(4096), dim3(256), 0, stream>>>(x, xh, 8192 * 1024);
  f32_to_f16_kernel<<<dim3(2048), dim3(256), 0, stream>>>(wi, wih, 3072 * 1024);
  f32_to_f16_kernel<<<dim3(1024), dim3(256), 0, stream>>>(wo, woh, 1024 * 1024);

  qkv_gemm_kernel<<<dim3(12, 256), dim3(256), 0, stream>>>(xh, wih, bi, qws, kws, vtws);

  const size_t lds_bytes =
      (size_t)(32 * 2048 + 256 + 32) * sizeof(float) + (size_t)256 * 64 * 2;
  attn_kernel<<<dim3(64, 64), dim3(256), lds_bytes, stream>>>(qws, kws, vtws, ctx, attn_w);

  out_gemm_kernel<<<dim3(4, 256), dim3(256), 0, stream>>>(ctx, woh, bo, attn_out);
}
